// ImportanceWeightedMoE_62732292325742
// MI455X (gfx1250) — matrix operand order repaired and checked by reading, not compiled, not run on hardware
//
#include <hip/hip_runtime.h>
#include <hip/hip_bf16.h>
#include <math.h>

// ---------------- types ----------------
typedef __attribute__((ext_vector_type(16))) __bf16 v16bf;
typedef __attribute__((ext_vector_type(8)))  __bf16 v8bf;
typedef __attribute__((ext_vector_type(4)))  __bf16 v4bf;
typedef __attribute__((ext_vector_type(8)))  float  v8f;

// ---------------- problem constants ----------------
#define NE   8
#define NB   8
#define NT   1024
#define TOK  8192      // NB*NT
#define DIN  640
#define DH   1024
#define DOUT 128
#define CODE 256

// output layout (floats): full_output, aux, diversity, ent[8], mean[8], std[8]
#define OUT_AUX    (8388608ull)
#define OUT_DIV    (8388609ull)
#define OUT_ENT    (8388610ull)
#define OUT_MEAN   (8388618ull)
#define OUT_STD    (8388626ull)

// ws layout: float region, then bf16 transposed weights
#define WS_IMP      0        // [8][640]   floats
#define WS_ANCH     5120     // [8][256]   floats
#define WS_EW       7168     // [8192][8]  floats
#define WS_ENTMEAN  72704    // [1]        float
#define WS_W1T_BYTE (524288ull)                     // bf16 [8][1024][640] = 10485760 B
#define WS_W2T_BYTE (524288ull + 10485760ull)       // bf16 [8][128][1024] =  2097152 B

// ---------------- wave helpers ----------------
__device__ __forceinline__ float wave_sum(float v) {
#pragma unroll
    for (int off = 16; off > 0; off >>= 1) v += __shfl_xor(v, off, 32);
    return v;
}
__device__ __forceinline__ float wave_max(float v) {
#pragma unroll
    for (int off = 16; off > 0; off >>= 1) v = fmaxf(v, __shfl_xor(v, off, 32));
    return v;
}

// ---------------- WMMA fragment loads ----------------
// A (16x32 bf16) from LDS: lane L: m=L&15, half=L>>4; elem j -> k = j + 16*half
__device__ __forceinline__ v16bf load_frag_a(const __bf16* p, int stride, int row_base,
                                             int k_base, int lane) {
    const int m = lane & 15, half = lane >> 4;
    const __bf16* q = p + (row_base + m) * stride + k_base + 16 * half;
    v8bf lo = *(const v8bf*)(q);
    v8bf hi = *(const v8bf*)(q + 8);
    v16bf r;
#pragma unroll
    for (int i = 0; i < 8; ++i) { r[i] = lo[i]; r[i + 8] = hi[i]; }
    return r;
}
// B (32x16 bf16) from GLOBAL, pre-transposed [n][k]: lane L: n=L&15, half=L>>4;
// elem j -> k = j + 16*half  (32 contiguous bytes per lane -> global_load_b128 x2)
__device__ __forceinline__ v16bf load_b_g(const __bf16* q) {
    v8bf lo = *(const v8bf*)(q);
    v8bf hi = *(const v8bf*)(q + 8);
    v16bf r;
#pragma unroll
    for (int i = 0; i < 8; ++i) { r[i] = lo[i]; r[i + 8] = hi[i]; }
    return r;
}

// =====================================================================
// Transpose f32 [E][R][C] -> bf16 [E][C][R]   (64x64 LDS tiles)
// =====================================================================
__global__ __launch_bounds__(256) void transpose_bf16_kernel(
    const float* __restrict__ src, __bf16* __restrict__ dst, int R, int C) {
    __shared__ __bf16 tile[64][72];
    const int e = blockIdx.z;
    const int r0 = blockIdx.x * 64;
    const int c0 = blockIdx.y * 64;
    const int t = threadIdx.x;
    const int tr = t >> 4;           // 0..15
    const int tc4 = (t & 15) * 4;    // 0..60
#pragma unroll
    for (int j = 0; j < 4; ++j) {
        const int r = tr + j * 16;
        float4 v = *(const float4*)(src + ((size_t)e * R + r0 + r) * C + c0 + tc4);
        tile[r][tc4 + 0] = (__bf16)v.x;
        tile[r][tc4 + 1] = (__bf16)v.y;
        tile[r][tc4 + 2] = (__bf16)v.z;
        tile[r][tc4 + 3] = (__bf16)v.w;
    }
    __syncthreads();
#pragma unroll
    for (int j = 0; j < 4; ++j) {
        const int c = tr + j * 16;   // output row (C index)
        v4bf o;
        o[0] = tile[tc4 + 0][c];
        o[1] = tile[tc4 + 1][c];
        o[2] = tile[tc4 + 2][c];
        o[3] = tile[tc4 + 3][c];
        *(v4bf*)(dst + ((size_t)e * C + c0 + c) * R + r0 + tc4) = o;
    }
}

// =====================================================================
// Kernel 1: importance softmax + stats, normalized anchors, diversity
// =====================================================================
__global__ __launch_bounds__(256) void setup_kernel(
    const float* __restrict__ code_anchor,  // [8,256]
    const float* __restrict__ fi,           // [8,640]
    float* __restrict__ ws_imp,             // [8,640]
    float* __restrict__ ws_anch,            // [8,256]
    float* __restrict__ ws_entmean,         // [1]
    float* __restrict__ out) {
    __shared__ float s_ent[NE], s_div[NE];
    const int lane = threadIdx.x & 31;
    const int e = threadIdx.x >> 5;  // one wave per expert

    float raw[20], ex[20];
    float mx = -1e30f;
#pragma unroll
    for (int j = 0; j < 20; ++j) {
        raw[j] = fi[e * DIN + lane + 32 * j];
        mx = fmaxf(mx, raw[j]);
    }
    mx = wave_max(mx);
    float ssum = 0.f;
#pragma unroll
    for (int j = 0; j < 20; ++j) { ex[j] = expf(2.f * raw[j] - 2.f * mx); ssum += ex[j]; }
    ssum = wave_sum(ssum);
    float ent = 0.f, sp = 0.f, sp2 = 0.f;
#pragma unroll
    for (int j = 0; j < 20; ++j) {
        float p = ex[j] / ssum;
        ws_imp[e * DIN + lane + 32 * j] = p;
        ent += -p * logf(p + 1e-8f);
        sp += p;
        sp2 += p * p;
    }
    ent = wave_sum(ent); sp = wave_sum(sp); sp2 = wave_sum(sp2);
    if (lane == 0) {
        out[OUT_ENT + e] = ent;
        out[OUT_MEAN + e] = sp / (float)DIN;
        float var = (sp2 - sp * sp / (float)DIN) / (float)(DIN - 1);
        out[OUT_STD + e] = sqrtf(fmaxf(var, 0.f));
        s_ent[e] = ent;
    }

    // normalized anchors
    float a[8]; float ss = 0.f;
#pragma unroll
    for (int j = 0; j < 8; ++j) { a[j] = code_anchor[e * CODE + lane + 32 * j]; ss += a[j] * a[j]; }
    ss = wave_sum(ss);
    float inv = 1.f / fmaxf(sqrtf(ss), 1e-12f);
#pragma unroll
    for (int j = 0; j < 8; ++j) ws_anch[e * CODE + lane + 32 * j] = a[j] * inv;

    // diversity
    float dsum = 0.f;
    for (int jj = 0; jj < NE; ++jj) {
        float d = 0.f;
#pragma unroll
        for (int j = 0; j < 20; ++j) d += raw[j] * fi[jj * DIN + lane + 32 * j];
        d = wave_sum(d);
        if (jj != e) dsum += d * d;
    }
    if (lane == 0) s_div[e] = dsum;
    __syncthreads();
    if (threadIdx.x == 0) {
        float tot = 0.f, em = 0.f;
#pragma unroll
        for (int k = 0; k < NE; ++k) { tot += s_div[k]; em += s_ent[k]; }
        out[OUT_DIV] = tot / (float)(NE * (NE - 1));
        ws_entmean[0] = em / (float)NE;
    }
}

// =====================================================================
// Kernel 2: router (cosine sim + gumbel softmax, tau=0.1) -> ws_ew
// =====================================================================
__global__ __launch_bounds__(256) void router_kernel(
    const float* __restrict__ code_emb,  // [8192,256]
    const float* __restrict__ gn,        // [8192,8]
    const float* __restrict__ anch,      // ws [8,256] normalized
    float* __restrict__ ew) {            // ws [8192,8]
    const int lane = threadIdx.x & 31;
    const int t = blockIdx.x * 8 + (threadIdx.x >> 5);

    float c[8]; float ss = 0.f;
#pragma unroll
    for (int j = 0; j < 8; ++j) {
        c[j] = code_emb[(size_t)t * CODE + lane + 32 * j];
        ss += c[j] * c[j];
    }
    ss = wave_sum(ss);
    const float inv = 1.f / fmaxf(sqrtf(ss), 1e-12f);

    float z[NE];
#pragma unroll
    for (int e = 0; e < NE; ++e) {
        float d = 0.f;
#pragma unroll
        for (int j = 0; j < 8; ++j) d += c[j] * anch[e * CODE + lane + 32 * j];
        d = wave_sum(d);
        z[e] = (d * inv + gn[(size_t)t * NE + e]) * 10.f;  // /tau
    }
    float mx = z[0];
#pragma unroll
    for (int e = 1; e < NE; ++e) mx = fmaxf(mx, z[e]);
    float se = 0.f;
#pragma unroll
    for (int e = 0; e < NE; ++e) { z[e] = expf(z[e] - mx); se += z[e]; }
    if (lane == 0) {
        const float r = 1.f / se;
#pragma unroll
        for (int e = 0; e < NE; ++e) ew[(size_t)t * NE + e] = z[e] * r;
    }
}

// =====================================================================
// Kernel 3: routing aux loss
// =====================================================================
__global__ __launch_bounds__(1024) void aux_kernel(
    const float* __restrict__ ew,        // ws [8192,8]
    const float* __restrict__ entmean,   // ws [1]
    float* __restrict__ out) {
    __shared__ float r1[32], r2[32];
    const int t = threadIdx.x;
    const int lane = t & 31, wv = t >> 5;

    float c[NE];
#pragma unroll
    for (int e = 0; e < NE; ++e) {
        float s = 0.f;
#pragma unroll
        for (int b = 0; b < NB; ++b) s += ew[((size_t)b * NT + t) * NE + e];
        c[e] = s;
    }
    float sc = 0.f, sc2 = 0.f;
#pragma unroll
    for (int e = 0; e < NE; ++e) { sc += c[e]; sc2 += c[e] * c[e]; }
    sc = wave_sum(sc); sc2 = wave_sum(sc2);
    if (lane == 0) { r1[wv] = sc; r2[wv] = sc2; }
    __syncthreads();
    if (wv == 0) {
        float a = r1[lane], b = r2[lane];
        a = wave_sum(a); b = wave_sum(b);
        if (lane == 0) { r1[0] = a; r2[0] = b; }
    }
    __syncthreads();
    const float S = r1[0], S2 = r2[0];
    float entp = 0.f;
#pragma unroll
    for (int e = 0; e < NE; ++e) {
        float l = c[e] / (S + 1e-8f);
        entp += -l * logf(l + 1e-8f);
    }
    __syncthreads();
    entp = wave_sum(entp);
    if (lane == 0) r1[wv] = entp;
    __syncthreads();
    if (wv == 0) {
        float a = r1[lane];
        a = wave_sum(a);
        if (lane == 0) {
            const float N = (float)(NT * NE);
            float var = (S2 - S * S / N) / (N - 1.f);
            float sd = sqrtf(fmaxf(var, 0.f));
            out[OUT_AUX] = 0.5f * (sd + a) + 0.01f * entmean[0];
        }
    }
}

// =====================================================================
// Kernel 4: fused importance-weighted 2-layer MLP + gating (bf16 WMMA)
//   - A fragments from LDS, B fragments direct from pre-transposed
//     bf16 weights in global (L2-resident) -> no barriers in K-loop
// =====================================================================
#define TM 64            // tokens per block
#define CHK 128          // DH chunk
#define XW_STRIDE 648    // DIN + 8
#define H_STRIDE  136    // CHK + 8

__global__ __launch_bounds__(256) void moe_mlp_kernel(
    const float*  __restrict__ h,    // [8192,640]
    const __bf16* __restrict__ W1T,  // ws [8][1024][640]
    const float*  __restrict__ b1,   // [8,1024]
    const __bf16* __restrict__ W2T,  // ws [8][128][1024]
    const float*  __restrict__ b2,   // [8,128]
    const float*  __restrict__ imp,  // ws [8,640]
    const float*  __restrict__ ew,   // ws [8192,8]
    float* __restrict__ out) {       // [8192,1024]
    __shared__ __bf16 s_xw[TM * XW_STRIDE];   // 82944 B
    __shared__ __bf16 s_hid[TM * H_STRIDE];   // 17408 B

    const int tid = threadIdx.x;
    const int lane = tid & 31;
    const int wv = tid >> 5;    // 8 waves
    const int wm = wv & 3;      // row group (4 x 16 rows)
    const int wn = wv >> 2;     // col group (2 x 64 cols)
    const int e = blockIdx.y;
    const int m0 = blockIdx.x * TM;
    const int half = lane >> 4;
    const int ln = lane & 15;

    // ---- stage Xw = h * importance[e], f32 -> bf16 (once per block) ----
    for (int i = tid; i < TM * (DIN / 4); i += 256) {
        const int m = i / (DIN / 4);
        const int kq = (i % (DIN / 4)) * 4;
        const float4 hv = *(const float4*)(h + (size_t)(m0 + m) * DIN + kq);
        const float4 iv = *(const float4*)(imp + e * DIN + kq);
        __bf16* dst = &s_xw[m * XW_STRIDE + kq];
        dst[0] = (__bf16)(hv.x * iv.x);
        dst[1] = (__bf16)(hv.y * iv.y);
        dst[2] = (__bf16)(hv.z * iv.z);
        dst[3] = (__bf16)(hv.w * iv.w);
    }
    __syncthreads();

    v8f oacc[4];
#pragma unroll
    for (int t = 0; t < 4; ++t)
#pragma unroll
        for (int i = 0; i < 8; ++i) oacc[t][i] = 0.f;

    for (int c = 0; c < DH / CHK; ++c) {
        const int dh0 = c * CHK;

        // ---- GEMM1: hid[:, dh0:dh0+128] = Xw @ W1 (no barriers inside) ----
        v8f acc1[4];
#pragma unroll
        for (int t = 0; t < 4; ++t)
#pragma unroll
            for (int i = 0; i < 8; ++i) acc1[t][i] = 0.f;

        const __bf16* wb[4];
#pragma unroll
        for (int tt = 0; tt < 4; ++tt)
            wb[tt] = W1T + ((size_t)e * DH + dh0 + wn * 64 + tt * 16 + ln) * DIN + 16 * half;

        for (int ks = 0; ks < DIN / 32; ++ks) {
            const v16bf A = load_frag_a(s_xw, XW_STRIDE, wm * 16, ks * 32, lane);
#pragma unroll
            for (int tt = 0; tt < 4; ++tt) {
                const v16bf B = load_b_g(wb[tt] + ks * 32);
                acc1[tt] = __builtin_amdgcn_wmma_f32_16x16x32_bf16(
                    false, A, false, B, (short)0, acc1[tt], false, false);
            }
        }

        // ---- bias + exact GELU -> s_hid (bf16) ----
#pragma unroll
        for (int tt = 0; tt < 4; ++tt) {
            const int n = wn * 64 + tt * 16 + ln;
            const float bb = b1[e * DH + dh0 + n];
#pragma unroll
            for (int v = 0; v < 8; ++v) {
                const float x = acc1[tt][v] + bb;
                const float g = 0.5f * x * (1.0f + erff(x * 0.70710678118654752f));
                s_hid[(wm * 16 + v + 8 * half) * H_STRIDE + n] = (__bf16)g;
            }
        }
        __syncthreads();

        // ---- GEMM2: oacc += hid @ W2[dh0:dh0+128, :] ----
#pragma unroll
        for (int ks2 = 0; ks2 < CHK / 32; ++ks2) {
            const v16bf A2 = load_frag_a(s_hid, H_STRIDE, wm * 16, ks2 * 32, lane);
#pragma unroll
            for (int tt = 0; tt < 4; ++tt) {
                const __bf16* q = W2T + ((size_t)e * DOUT + wn * 64 + tt * 16 + ln) * DH +
                                  dh0 + ks2 * 32 + 16 * half;
                oacc[tt] = __builtin_amdgcn_wmma_f32_16x16x32_bf16(
                    false, A2, false, load_b_g(q), (short)0, oacc[tt], false, false);
            }
        }
        __syncthreads();  // protect s_hid before next chunk rewrites
    }

    // ---- gate by expert weight, add b2, store output slice ----
    float ewv[8];
#pragma unroll
    for (int v = 0; v < 8; ++v)
        ewv[v] = ew[(size_t)(m0 + wm * 16 + v + 8 * half) * NE + e];
#pragma unroll
    for (int tt = 0; tt < 4; ++tt) {
        const int n = wn * 64 + tt * 16 + ln;
        const float b2v = b2[e * DOUT + n];
#pragma unroll
        for (int v = 0; v < 8; ++v) {
            const int tok = m0 + wm * 16 + v + 8 * half;
            out[(size_t)tok * (NE * DOUT) + e * DOUT + n] = (oacc[tt][v] + b2v) * ewv[v];
        }
    }
}

// =====================================================================
// launcher
// =====================================================================
extern "C" void kernel_launch(void* const* d_in, const int* in_sizes, int n_in,
                              void* d_out, int out_size, void* d_ws, size_t ws_size,
                              hipStream_t stream) {
    const float* h        = (const float*)d_in[0];
    const float* code_emb = (const float*)d_in[1];
    const float* gn       = (const float*)d_in[2];
    const float* ca       = (const float*)d_in[3];
    const float* fi       = (const float*)d_in[4];
    const float* W1       = (const float*)d_in[5];
    const float* b1       = (const float*)d_in[6];
    const float* W2       = (const float*)d_in[7];
    const float* b2       = (const float*)d_in[8];
    float* out = (float*)d_out;
    float* ws  = (float*)d_ws;

    float*  ws_imp     = ws + WS_IMP;
    float*  ws_anch    = ws + WS_ANCH;
    float*  ws_ew      = ws + WS_EW;
    float*  ws_entmean = ws + WS_ENTMEAN;
    __bf16* w1t = (__bf16*)((char*)d_ws + WS_W1T_BYTE);
    __bf16* w2t = (__bf16*)((char*)d_ws + WS_W2T_BYTE);

    setup_kernel<<<1, 256, 0, stream>>>(ca, fi, ws_imp, ws_anch, ws_entmean, out);
    router_kernel<<<TOK / 8, 256, 0, stream>>>(code_emb, gn, ws_anch, ws_ew);
    aux_kernel<<<1, 1024, 0, stream>>>(ws_ew, ws_entmean, out);

    transpose_bf16_kernel<<<dim3(DIN / 64, DH / 64, NE), 256, 0, stream>>>(W1, w1t, DIN, DH);
    transpose_bf16_kernel<<<dim3(DH / 64, DOUT / 64, NE), 256, 0, stream>>>(W2, w2t, DH, DOUT);

    dim3 grid(TOK / TM, NE);
    moe_mlp_kernel<<<grid, 256, 0, stream>>>(h, w1t, b1, w2t, b2, ws_imp, ws_ew, out);
}
